// TemporalAttention_1219770712668
// MI455X (gfx1250) — matrix operand order repaired and checked by reading, not compiled, not run on hardware
//
#include <hip/hip_runtime.h>

typedef __attribute__((ext_vector_type(16))) __bf16 bf16x16;
typedef __attribute__((ext_vector_type(8)))  float  v8f;
typedef __attribute__((ext_vector_type(4)))  float  f32x4;
typedef __attribute__((ext_vector_type(4)))  unsigned int u32x4;

union AFrag { bf16x16 v; u32x4 q[2]; };

static __device__ inline v8f wmma_bf16(bf16x16 a, bf16x16 b, v8f c) {
  return __builtin_amdgcn_wmma_f32_16x16x32_bf16(
      /*neg_a=*/false, a, /*neg_b=*/false, b,
      /*c_mod=*/(short)0, c, /*reuse_a=*/false, /*reuse_b=*/false);
}

// A matrix fragment (16x32, bf16, row-major, lda elements).
// ISA layout: lane<16 -> row=lane, v[0..7]=K(0..7), v[8..15]=K(16..23)
//             lane>=16 -> row=lane-16, v[0..7]=K(8..15), v[8..15]=K(24..31)
static __device__ inline bf16x16 load_a_bf16(const __bf16* A, long lda, int k0, int lane) {
  int m  = lane & 15;
  int kh = (lane >> 4) * 8;
  const __bf16* p = A + (long)m * lda + k0 + kh;
  AFrag f;
  f.q[0] = *reinterpret_cast<const u32x4*>(p);        // K offsets +0..7
  f.q[1] = *reinterpret_cast<const u32x4*>(p + 16);   // K offsets +16..23
  return f.v;
}

// Same fragment but source is fp32 (converted on the fly).
static __device__ inline bf16x16 load_a_f32(const float* A, long lda, int k0, int lane) {
  int m  = lane & 15;
  int kh = (lane >> 4) * 8;
  const float* p = A + (long)m * lda + k0 + kh;
  f32x4 x0 = *reinterpret_cast<const f32x4*>(p);
  f32x4 x1 = *reinterpret_cast<const f32x4*>(p + 4);
  f32x4 y0 = *reinterpret_cast<const f32x4*>(p + 16);
  f32x4 y1 = *reinterpret_cast<const f32x4*>(p + 20);
  bf16x16 v;
  v[0]=(__bf16)x0.x; v[1]=(__bf16)x0.y; v[2]=(__bf16)x0.z; v[3]=(__bf16)x0.w;
  v[4]=(__bf16)x1.x; v[5]=(__bf16)x1.y; v[6]=(__bf16)x1.z; v[7]=(__bf16)x1.w;
  v[8]=(__bf16)y0.x; v[9]=(__bf16)y0.y; v[10]=(__bf16)y0.z; v[11]=(__bf16)y0.w;
  v[12]=(__bf16)y1.x; v[13]=(__bf16)y1.y; v[14]=(__bf16)y1.z; v[15]=(__bf16)y1.w;
  return v;
}

// B matrix fragment (32x16 K-major math view), sourced from BT stored N x K row-major.
// ISA layout: lane<16 -> col=lane, v[0..7]=K(0..7), v[8..15]=K(16..23); lane>=16 -> col=lane-16, v[0..7]=K(8..15), v[8..15]=K(24..31)
static __device__ inline bf16x16 load_b_bf16(const __bf16* BT, long ldb, int k0, int lane) {
  int n  = lane & 15;
  int kh = (lane >> 4) * 8;
  const __bf16* p = BT + (long)n * ldb + k0 + kh;
  AFrag f;
  f.q[0] = *reinterpret_cast<const u32x4*>(p);
  f.q[1] = *reinterpret_cast<const u32x4*>(p + 16);
  return f.v;
}

static __device__ inline v8f zero8() { v8f z = {0.f,0.f,0.f,0.f,0.f,0.f,0.f,0.f}; return z; }

// ---------------------------------------------------------------------------
// elementwise fp32 -> bf16
__global__ void cvt_f32_bf16_kernel(const float* __restrict__ src, __bf16* __restrict__ dst, long n) {
  long i = (long)blockIdx.x * blockDim.x + threadIdx.x;
  long stride = (long)gridDim.x * blockDim.x;
  for (; i < n; i += stride) dst[i] = (__bf16)src[i];
}

// rel_k (257x64) -> RK bf16 padded 272x64 ; rel_v (257x64) -> RVT bf16 transposed padded 64x288
__global__ void build_rel_kernel(const float* __restrict__ rel_k, const float* __restrict__ rel_v,
                                 __bf16* __restrict__ RK, __bf16* __restrict__ RVT) {
  int i = blockIdx.x * blockDim.x + threadIdx.x;
  if (i < 272 * 64) {
    int j = i >> 6, d = i & 63;
    RK[i] = (__bf16)((j < 257) ? rel_k[j * 64 + d] : 0.f);
  }
  if (i < 64 * 288) {
    int d = i / 288, j = i % 288;
    RVT[i] = (__bf16)((j < 257) ? rel_v[j * 64 + d] : 0.f);
  }
}

// ---------------------------------------------------------------------------
// Projection: OUT_headed = X[8192x1024] @ W^T (W row-major 1024x1024 == BT layout)
// mode 0: OUT[((b*16+h)*1024+s)*64+d]   (Q/K layout)
// mode 1: OUT[((b*16+h)*64+d)*1024+s]   (V transposed layout)
__global__ __launch_bounds__(128) void proj_kernel(const __bf16* __restrict__ X,
                                                   const __bf16* __restrict__ W,
                                                   __bf16* __restrict__ OUT, int mode) {
  const int K = 1024;
  int lane = threadIdx.x, wave = threadIdx.y;
  int m0 = blockIdx.y * 64 + (wave & 1) * 32;
  int n0 = blockIdx.x * 64 + (wave >> 1) * 32;
  const __bf16* A  = X + (long)m0 * K;
  const __bf16* BT = W + (long)n0 * K;
  v8f acc[4] = {zero8(), zero8(), zero8(), zero8()};
  for (int k0 = 0; k0 < K; k0 += 32) {
    bf16x16 a0 = load_a_bf16(A,           K, k0, lane);
    bf16x16 a1 = load_a_bf16(A + 16 * K,  K, k0, lane);
    bf16x16 b0 = load_b_bf16(BT,          K, k0, lane);
    bf16x16 b1 = load_b_bf16(BT + 16 * K, K, k0, lane);
    acc[0] = wmma_bf16(a0, b0, acc[0]);
    acc[1] = wmma_bf16(a0, b1, acc[1]);
    acc[2] = wmma_bf16(a1, b0, acc[2]);
    acc[3] = wmma_bf16(a1, b1, acc[3]);
  }
  int cn = lane & 15, rh = lane >> 4;
  for (int i = 0; i < 2; ++i)
    for (int j = 0; j < 2; ++j)
      for (int v = 0; v < 8; ++v) {
        int row = m0 + i * 16 + v + 8 * rh;           // (b,s)
        int col = n0 + j * 16 + cn;                   // (h,d)
        int b = row >> 10, s = row & 1023;
        int h = col >> 6,  d = col & 63;
        long idx = (mode == 0)
                     ? ((((long)b * 16 + h) * 1024 + s) * 64 + d)
                     : ((((long)b * 16 + h) * 64 + d) * 1024 + s);
        OUT[idx] = (__bf16)acc[i * 2 + j][v];
      }
}

// P[131072 x 272] = Qh[131072 x 64] @ RK^T (RK is 272x64 row-major == BT)
__global__ __launch_bounds__(128) void gemm_p_kernel(const __bf16* __restrict__ Qh,
                                                     const __bf16* __restrict__ RK,
                                                     float* __restrict__ P) {
  const int N = 272;
  int lane = threadIdx.x, wave = threadIdx.y;
  long m0 = (long)blockIdx.y * 64 + (wave & 1) * 32;
  int  n0 = blockIdx.x * 64 + (wave >> 1) * 32;
  if (n0 >= N) return;                    // wave-uniform
  bool nv1 = (n0 + 16) < N;
  const __bf16* A   = Qh + m0 * 64;
  const __bf16* BT0 = RK + (long)n0 * 64;
  v8f acc[4] = {zero8(), zero8(), zero8(), zero8()};
  for (int k0 = 0; k0 < 64; k0 += 32) {
    bf16x16 a0 = load_a_bf16(A,           64, k0, lane);
    bf16x16 a1 = load_a_bf16(A + 16 * 64, 64, k0, lane);
    bf16x16 b0 = load_b_bf16(BT0,         64, k0, lane);
    acc[0] = wmma_bf16(a0, b0, acc[0]);
    acc[2] = wmma_bf16(a1, b0, acc[2]);
    if (nv1) {
      bf16x16 b1 = load_b_bf16(BT0 + 16 * 64, 64, k0, lane);
      acc[1] = wmma_bf16(a0, b1, acc[1]);
      acc[3] = wmma_bf16(a1, b1, acc[3]);
    }
  }
  int cn = lane & 15, rh = lane >> 4;
  for (int i = 0; i < 2; ++i)
    for (int j = 0; j < 2; ++j) {
      if (j == 1 && !nv1) continue;
      for (int v = 0; v < 8; ++v) {
        long row = m0 + i * 16 + v + 8 * rh;
        int  col = n0 + j * 16 + cn;
        P[row * 272 + col] = acc[i * 2 + j][v];
      }
    }
}

// Scores + bias + softmax + attn write + A2 bucket sums. One wave per (b,h, 16 query rows).
// LDS: 16 rows x 1024 fp32 = 64 KB (dynamic).
__global__ __launch_bounds__(32) void scores_softmax_kernel(const __bf16* __restrict__ Qh,
                                                            const __bf16* __restrict__ Kh,
                                                            const float* __restrict__ P,
                                                            const float* __restrict__ decay,
                                                            float* __restrict__ attn,
                                                            __bf16* __restrict__ A2) {
  extern __shared__ float smem[];
  const int STRIDE = 1024;
  int lane = threadIdx.x;
  int z = blockIdx.y;                 // b*16 + h
  int h = z & 15;
  int l0 = blockIdx.x * 16;
  const __bf16* Aq = Qh + ((long)z * 1024 + l0) * 64;
  bf16x16 a0 = load_a_bf16(Aq, 64, 0, lane);
  bf16x16 a1 = load_a_bf16(Aq, 64, 32, lane);
  float dec = decay[h];
  int cn = lane & 15, rh = lane >> 4;
  const float invs = 0.125f;          // 1/sqrt(64)

  for (int nt = 0; nt < 64; ++nt) {
    int r0 = nt * 16;
    const __bf16* Bk = Kh + ((long)z * 1024 + r0) * 64;
    bf16x16 b0 = load_b_bf16(Bk, 64, 0, lane);
    bf16x16 b1 = load_b_bf16(Bk, 64, 32, lane);
    v8f acc = zero8();
    acc = wmma_bf16(a0, b0, acc);
    acc = wmma_bf16(a1, b1, acc);
    for (int v = 0; v < 8; ++v) {
      int m = v + 8 * rh;
      int l = l0 + m;
      int r = r0 + cn;
      int dlt = r - l;
      int dc  = dlt < -128 ? -128 : (dlt > 128 ? 128 : dlt);
      float val = acc[v] * invs
                + P[((long)z * 1024 + l) * 272 + (dc + 128)]
                - dec * fabsf((float)dlt);
      smem[m * STRIDE + r] = val;
    }
  }
  __syncthreads();

  int row = lane & 15, half = lane >> 4;
  float* rp = smem + row * STRIDE;
  int c0 = half * 512;
  float mx = -3.4e38f;
  for (int c = c0; c < c0 + 512; ++c) mx = fmaxf(mx, rp[c]);
  mx = fmaxf(mx, __shfl_xor(mx, 16, 32));
  float sum = 0.f;
  for (int c = c0; c < c0 + 512; ++c) { float e = __expf(rp[c] - mx); rp[c] = e; sum += e; }
  sum += __shfl_xor(sum, 16, 32);
  float inv = 1.0f / sum;
  float* orow = attn + ((long)z * 1024 + l0 + row) * 1024;
  for (int c = c0; c < c0 + 512; c += 4) {
    f32x4 e = *reinterpret_cast<f32x4*>(rp + c);
    e *= inv;
    *reinterpret_cast<f32x4*>(rp + c) = e;
    *reinterpret_cast<f32x4*>(orow + c) = e;
  }
  __syncthreads();

  // A2 bucket sums: j in [0,256], zero-pad to 288 (K multiple of 32)
  int l = l0 + row;
  __bf16* a2p = A2 + ((long)z * 1024 + l) * 288;
  if (half == 0) {
    float s0 = 0.f;
    for (int r = 0; r <= l - 128; ++r) s0 += rp[r];
    a2p[0] = (__bf16)s0;
    for (int j = 1; j < 128; ++j) {
      int r = l + j - 128;
      a2p[j] = (__bf16)((r >= 0) ? rp[r] : 0.f);
    }
  } else {
    for (int j = 128; j < 256; ++j) {
      int r = l + j - 128;
      a2p[j] = (__bf16)((r < 1024) ? rp[r] : 0.f);
    }
    float s1 = 0.f;
    for (int r = l + 128; r < 1024; ++r) s1 += rp[r];
    a2p[256] = (__bf16)s1;
    for (int j = 257; j < 288; ++j) a2p[j] = (__bf16)0.f;
  }
}

// ctx[b,s,h*64+d] = attn[z] (fp32, converted on load) @ V  (Vt is N x K row-major)
__global__ __launch_bounds__(128) void attn_av_kernel(const float* __restrict__ attn,
                                                      const __bf16* __restrict__ Vt,
                                                      float* __restrict__ ctx) {
  int z = blockIdx.z;
  int lane = threadIdx.x, wave = threadIdx.y;
  int m0 = blockIdx.y * 64 + (wave & 1) * 32;
  int n0 = (wave >> 1) * 32;
  const float*  A  = attn + (long)z * 1024 * 1024 + (long)m0 * 1024;
  const __bf16* BT = Vt   + (long)z * 64 * 1024   + (long)n0 * 1024;
  v8f acc[4] = {zero8(), zero8(), zero8(), zero8()};
  for (int k0 = 0; k0 < 1024; k0 += 32) {
    bf16x16 a0 = load_a_f32(A,             1024, k0, lane);
    bf16x16 a1 = load_a_f32(A + 16 * 1024, 1024, k0, lane);
    bf16x16 b0 = load_b_bf16(BT,             1024, k0, lane);
    bf16x16 b1 = load_b_bf16(BT + 16 * 1024, 1024, k0, lane);
    acc[0] = wmma_bf16(a0, b0, acc[0]);
    acc[1] = wmma_bf16(a0, b1, acc[1]);
    acc[2] = wmma_bf16(a1, b0, acc[2]);
    acc[3] = wmma_bf16(a1, b1, acc[3]);
  }
  int b = z >> 4, h = z & 15;
  float* C = ctx + (long)b * 1024 * 1024 + h * 64;
  int cn = lane & 15, rh = lane >> 4;
  for (int i = 0; i < 2; ++i)
    for (int j = 0; j < 2; ++j)
      for (int v = 0; v < 8; ++v)
        C[(long)(m0 + i * 16 + v + 8 * rh) * 1024 + (n0 + j * 16 + cn)] = acc[i * 2 + j][v];
}

// ctx += A2[z] (1024x288 bf16) @ rel_v  (RVT is 64x288 row-major)
__global__ __launch_bounds__(128) void gemm_a2_kernel(const __bf16* __restrict__ A2,
                                                      const __bf16* __restrict__ RVT,
                                                      float* __restrict__ ctx) {
  int z = blockIdx.z;
  int lane = threadIdx.x, wave = threadIdx.y;
  int m0 = blockIdx.y * 64 + (wave & 1) * 32;
  int n0 = (wave >> 1) * 32;
  const __bf16* A  = A2  + (long)z * 1024 * 288 + (long)m0 * 288;
  const __bf16* BT = RVT + (long)n0 * 288;
  v8f acc[4] = {zero8(), zero8(), zero8(), zero8()};
  for (int k0 = 0; k0 < 288; k0 += 32) {
    bf16x16 a0 = load_a_bf16(A,            288, k0, lane);
    bf16x16 a1 = load_a_bf16(A + 16 * 288, 288, k0, lane);
    bf16x16 b0 = load_b_bf16(BT,            288, k0, lane);
    bf16x16 b1 = load_b_bf16(BT + 16 * 288, 288, k0, lane);
    acc[0] = wmma_bf16(a0, b0, acc[0]);
    acc[1] = wmma_bf16(a0, b1, acc[1]);
    acc[2] = wmma_bf16(a1, b0, acc[2]);
    acc[3] = wmma_bf16(a1, b1, acc[3]);
  }
  int b = z >> 4, h = z & 15;
  float* C = ctx + (long)b * 1024 * 1024 + h * 64;
  int cn = lane & 15, rh = lane >> 4;
  for (int i = 0; i < 2; ++i)
    for (int j = 0; j < 2; ++j)
      for (int v = 0; v < 8; ++v) {
        long idx = (long)(m0 + i * 16 + v + 8 * rh) * 1024 + (n0 + j * 16 + cn);
        C[idx] += acc[i * 2 + j][v];
      }
}

// out[8192x1024] = ctxbf @ w_o^T + b_o
__global__ __launch_bounds__(128) void gemm_out_kernel(const __bf16* __restrict__ X,
                                                       const __bf16* __restrict__ W,
                                                       const float* __restrict__ bias,
                                                       float* __restrict__ OUT) {
  const int K = 1024;
  int lane = threadIdx.x, wave = threadIdx.y;
  int m0 = blockIdx.y * 64 + (wave & 1) * 32;
  int n0 = blockIdx.x * 64 + (wave >> 1) * 32;
  const __bf16* A  = X + (long)m0 * K;
  const __bf16* BT = W + (long)n0 * K;
  v8f acc[4] = {zero8(), zero8(), zero8(), zero8()};
  for (int k0 = 0; k0 < K; k0 += 32) {
    bf16x16 a0 = load_a_bf16(A,           K, k0, lane);
    bf16x16 a1 = load_a_bf16(A + 16 * K,  K, k0, lane);
    bf16x16 b0 = load_b_bf16(BT,          K, k0, lane);
    bf16x16 b1 = load_b_bf16(BT + 16 * K, K, k0, lane);
    acc[0] = wmma_bf16(a0, b0, acc[0]);
    acc[1] = wmma_bf16(a0, b1, acc[1]);
    acc[2] = wmma_bf16(a1, b0, acc[2]);
    acc[3] = wmma_bf16(a1, b1, acc[3]);
  }
  int cn = lane & 15, rh = lane >> 4;
  for (int i = 0; i < 2; ++i)
    for (int j = 0; j < 2; ++j)
      for (int v = 0; v < 8; ++v) {
        int row = m0 + i * 16 + v + 8 * rh;
        int col = n0 + j * 16 + cn;
        OUT[(long)row * 1024 + col] = acc[i * 2 + j][v] + bias[col];
      }
}

// ---------------------------------------------------------------------------
extern "C" void kernel_launch(void* const* d_in, const int* in_sizes, int n_in,
                              void* d_out, int out_size, void* d_ws, size_t ws_size,
                              hipStream_t stream) {
  const float* q_in  = (const float*)d_in[0];
  const float* k_in  = (const float*)d_in[1];
  const float* v_in  = (const float*)d_in[2];
  const float* w_q   = (const float*)d_in[3];
  const float* w_k   = (const float*)d_in[4];
  const float* w_v   = (const float*)d_in[5];
  const float* w_o   = (const float*)d_in[6];
  const float* b_o   = (const float*)d_in[7];
  const float* rel_k = (const float*)d_in[8];
  const float* rel_v = (const float*)d_in[9];
  const float* decay = (const float*)d_in[10];

  float* out  = (float*)d_out;
  float* attn = out + (size_t)8 * 1024 * 1024;   // [8,16,1024,1024]

  char* ws = (char*)d_ws;
  size_t off = 0;
  auto take = [&](size_t bytes) -> void* {
    void* p = ws + off;
    off += (bytes + 255) & ~(size_t)255;
    return p;
  };
  const size_t BS = 8192, D = 1024;
  __bf16* xq  = (__bf16*)take(BS * D * 2);
  __bf16* xk  = (__bf16*)take(BS * D * 2);
  __bf16* xv  = (__bf16*)take(BS * D * 2);
  __bf16* wqb = (__bf16*)take(D * D * 2);
  __bf16* wkb = (__bf16*)take(D * D * 2);
  __bf16* wvb = (__bf16*)take(D * D * 2);
  __bf16* wob = (__bf16*)take(D * D * 2);
  __bf16* RK  = (__bf16*)take(272 * 64 * 2);
  __bf16* RVT = (__bf16*)take(64 * 288 * 2);
  __bf16* Qh  = (__bf16*)take((size_t)128 * 1024 * 64 * 2);
  __bf16* Kh  = (__bf16*)take((size_t)128 * 1024 * 64 * 2);
  __bf16* Vt  = (__bf16*)take((size_t)128 * 64 * 1024 * 2);
  float*  P   = (float*) take((size_t)131072 * 272 * 4);
  __bf16* A2  = (__bf16*)take((size_t)131072 * 288 * 2);
  float*  ctx = (float*) take(BS * D * 4);
  __bf16* ctb = (__bf16*)take(BS * D * 2);

  // 1) fp32 -> bf16 conversions
  cvt_f32_bf16_kernel<<<dim3(4096), dim3(256), 0, stream>>>(q_in, xq, (long)BS * D);
  cvt_f32_bf16_kernel<<<dim3(4096), dim3(256), 0, stream>>>(k_in, xk, (long)BS * D);
  cvt_f32_bf16_kernel<<<dim3(4096), dim3(256), 0, stream>>>(v_in, xv, (long)BS * D);
  cvt_f32_bf16_kernel<<<dim3(1024), dim3(256), 0, stream>>>(w_q, wqb, (long)D * D);
  cvt_f32_bf16_kernel<<<dim3(1024), dim3(256), 0, stream>>>(w_k, wkb, (long)D * D);
  cvt_f32_bf16_kernel<<<dim3(1024), dim3(256), 0, stream>>>(w_v, wvb, (long)D * D);
  cvt_f32_bf16_kernel<<<dim3(1024), dim3(256), 0, stream>>>(w_o, wob, (long)D * D);
  build_rel_kernel<<<dim3(72), dim3(256), 0, stream>>>(rel_k, rel_v, RK, RVT);

  // 2) projections (WMMA bf16): Q/K headed layout, V transposed layout
  proj_kernel<<<dim3(16, 128), dim3(32, 4), 0, stream>>>(xq, wqb, Qh, 0);
  proj_kernel<<<dim3(16, 128), dim3(32, 4), 0, stream>>>(xk, wkb, Kh, 0);
  proj_kernel<<<dim3(16, 128), dim3(32, 4), 0, stream>>>(xv, wvb, Vt, 1);

  // 3) P = Q @ rel_k^T  [131072 x 272]
  gemm_p_kernel<<<dim3(5, 2048), dim3(32, 4), 0, stream>>>(Qh, RK, P);

  // 4) scores + bias + softmax + attn output + A2 buckets (64 KB LDS per wave-group)
  scores_softmax_kernel<<<dim3(64, 128), dim3(32), 16 * 1024 * 4, stream>>>(
      Qh, Kh, P, decay, attn, A2);

  // 5) ctx = attn @ V ; ctx += A2 @ rel_v
  attn_av_kernel<<<dim3(1, 16, 128), dim3(32, 4), 0, stream>>>(attn, Vt, ctx);
  gemm_a2_kernel<<<dim3(1, 16, 128), dim3(32, 4), 0, stream>>>(A2, RVT, ctx);

  // 6) output projection
  cvt_f32_bf16_kernel<<<dim3(4096), dim3(256), 0, stream>>>(ctx, ctb, (long)BS * D);
  gemm_out_kernel<<<dim3(16, 128), dim3(32, 4), 0, stream>>>(ctb, wob, b_o, out);
}
